// Qwen3MoeSparseMoeBlock_32495722561889
// MI455X (gfx1250) — matrix operand order repaired and checked by reading, not compiled, not run on hardware
//
#include <hip/hip_runtime.h>
#include <cstdint>
#include <cstddef>

#define NUM_EXPERTS 64
#define TOPK 8
#define HIDDEN 2048
#define MLP 768
#define TOKENS 1024

typedef __attribute__((ext_vector_type(16))) __bf16 v16bf;
typedef __attribute__((ext_vector_type(2)))  __bf16 bf16x2;
typedef __attribute__((ext_vector_type(8)))  float  v8f;

// Native f32->bf16 (RTNE) — lowers to v_cvt_pk_bf16_f32 / v_cvt_bf16_f32 on
// gfx1250 instead of the ~6-VALU software rounding sequence.
__device__ __forceinline__ unsigned pack2(float a, float b) {
    bf16x2 v = { (__bf16)a, (__bf16)b };
    return __builtin_bit_cast(unsigned, v);
}
__device__ __forceinline__ unsigned short f2bf(float f) {
    return __builtin_bit_cast(unsigned short, (__bf16)f);
}

// ---------------------------------------------------------------------------
// K1: router logits + top-8 + softmax + per-expert counts. One wave per token.
// ---------------------------------------------------------------------------
__global__ __launch_bounds__(256) void moe_router(
    const float* __restrict__ hs, const float* __restrict__ rw,
    int* __restrict__ selIdx, float* __restrict__ selW, int* __restrict__ counts) {
    int wid = threadIdx.x >> 5, lane = threadIdx.x & 31;
    int t = blockIdx.x * 8 + wid;
    if (t >= TOKENS) return;
    const float* hrow = hs + (size_t)t * HIDDEN;
    float a0 = 0.f, a1 = 0.f;
    #pragma unroll 4
    for (int h = 0; h < HIDDEN; ++h) {
        float x = hrow[h];
        a0 = fmaf(x, rw[h * NUM_EXPERTS + lane], a0);
        a1 = fmaf(x, rw[h * NUM_EXPERTS + lane + 32], a1);
    }
    float topv[TOPK]; int topi[TOPK];
    #pragma unroll
    for (int k = 0; k < TOPK; ++k) {
        float m; int mi;
        if (a0 >= a1) { m = a0; mi = lane; } else { m = a1; mi = lane + 32; }
        #pragma unroll
        for (int off = 16; off >= 1; off >>= 1) {
            float om = __shfl_xor(m, off, 32);
            int   oi = __shfl_xor(mi, off, 32);
            if (om > m || (om == m && oi < mi)) { m = om; mi = oi; }
        }
        topv[k] = m; topi[k] = mi;
        if (mi == lane)           a0 = -__builtin_inff();
        else if (mi == lane + 32) a1 = -__builtin_inff();
    }
    float mx = topv[0], s = 0.f, w[TOPK];
    #pragma unroll
    for (int k = 0; k < TOPK; ++k) { w[k] = __expf(topv[k] - mx); s += w[k]; }
    float inv = __builtin_amdgcn_rcpf(s);
    if (lane == 0) {
        for (int k = 0; k < TOPK; ++k) {
            selIdx[t * TOPK + k] = topi[k];
            selW[t * TOPK + k]   = w[k] * inv;
            atomicAdd(&counts[topi[k]], 1);
        }
    }
}

// ---------------------------------------------------------------------------
// K2: exclusive scan of 64 counts -> offsets, cursors
// ---------------------------------------------------------------------------
__global__ void moe_scan(const int* __restrict__ counts,
                         int* __restrict__ offsets, int* __restrict__ cursor) {
    if (threadIdx.x == 0 && blockIdx.x == 0) {
        int acc = 0;
        for (int e = 0; e < NUM_EXPERTS; ++e) {
            offsets[e] = acc; cursor[e] = acc; acc += counts[e];
        }
    }
}

// ---------------------------------------------------------------------------
// K3: scatter (token,k) into expert-sorted row order
// ---------------------------------------------------------------------------
__global__ __launch_bounds__(256) void moe_scatter(
    const int* __restrict__ selIdx, const float* __restrict__ selW,
    int* __restrict__ cursor, int* __restrict__ rowTok, float* __restrict__ rowW) {
    int t = blockIdx.x * blockDim.x + threadIdx.x;
    if (t >= TOKENS) return;
    for (int k = 0; k < TOPK; ++k) {
        int e = selIdx[t * TOPK + k];
        int pos = atomicAdd(&cursor[e], 1);
        rowTok[pos] = t;
        rowW[pos]   = selW[t * TOPK + k];
    }
}

// ---------------------------------------------------------------------------
// K4: grouped gate/up GEMM (bf16 WMMA), double-buffered LDS, one barrier/step.
//     Block tile: M=32, N=64, K-step=32. 8 waves: 2(M) x 4(N), 16x16 each.
//     LDS holds bf16 packed pairs (k even = low half) matching WMMA fragments.
// ---------------------------------------------------------------------------
__global__ __launch_bounds__(256) void moe_gateup(
    const float* __restrict__ hs, const float* __restrict__ gateW,
    const float* __restrict__ upW, const int* __restrict__ offsets,
    const int* __restrict__ counts, const int* __restrict__ rowTok,
    const float* __restrict__ rowW, unsigned short* __restrict__ act) {
    __shared__ __attribute__((aligned(16))) unsigned ldsA [2][32][20]; // [row][kpair]
    __shared__ __attribute__((aligned(16))) unsigned ldsBg[2][64][20]; // [n][kpair]
    __shared__ __attribute__((aligned(16))) unsigned ldsBu[2][64][20];

    int e = blockIdx.y;
    int count = counts[e];
    if (count == 0) return;
    int base = offsets[e];
    int n0 = blockIdx.x * 64;

    int tid = threadIdx.x, lane = tid & 31, wid = tid >> 5;
    int wm = wid & 1, wn = wid >> 1;
    int h = lane >> 4, l15 = lane & 15;
    int arow = tid >> 3, akq = (tid & 7) * 4;   // A: row, float offset in K-tile
    int kp = tid >> 4, n4 = (tid & 15) * 4;     // B: k-pair index, n offset
    const int NT = HIDDEN / 32;

    for (int mt = 0; mt * 32 < count; ++mt) {
        int aidx = mt * 32 + arow; if (aidx >= count) aidx = count - 1;
        const float* aptr = hs + (size_t)rowTok[base + aidx] * HIDDEN + akq;
        const float* gptr = gateW + ((size_t)e * HIDDEN + 2 * kp) * MLP + n0 + n4;
        const float* uptr = upW   + ((size_t)e * HIDDEN + 2 * kp) * MLP + n0 + n4;

        // prologue: registers for tile 0
        float4 av = *(const float4*)(aptr);
        float4 g0 = *(const float4*)(gptr);
        float4 g1 = *(const float4*)(gptr + MLP);
        float4 u0 = *(const float4*)(uptr);
        float4 u1 = *(const float4*)(uptr + MLP);

        v8f accG = {}; v8f accU = {};
        for (int it = 0; it < NT; ++it) {
            unsigned (*pA)[20] = ldsA [it & 1];
            unsigned (*pG)[20] = ldsBg[it & 1];
            unsigned (*pU)[20] = ldsBu[it & 1];
            pA[arow][(akq >> 1) + 0] = pack2(av.x, av.y);
            pA[arow][(akq >> 1) + 1] = pack2(av.z, av.w);
            pG[n4 + 0][kp] = pack2(g0.x, g1.x);
            pG[n4 + 1][kp] = pack2(g0.y, g1.y);
            pG[n4 + 2][kp] = pack2(g0.z, g1.z);
            pG[n4 + 3][kp] = pack2(g0.w, g1.w);
            pU[n4 + 0][kp] = pack2(u0.x, u1.x);
            pU[n4 + 1][kp] = pack2(u0.y, u1.y);
            pU[n4 + 2][kp] = pack2(u0.z, u1.z);
            pU[n4 + 3][kp] = pack2(u0.w, u1.w);

            if (it + 1 < NT) {                      // stage next tile into regs
                size_t ko = (size_t)(it + 1) * 32;
                av = *(const float4*)(aptr + ko);
                g0 = *(const float4*)(gptr + ko * MLP);
                g1 = *(const float4*)(gptr + ko * MLP + MLP);
                u0 = *(const float4*)(uptr + ko * MLP);
                u1 = *(const float4*)(uptr + ko * MLP + MLP);
            }
            if (it + 2 < NT) {                      // keep weight streams hot
                size_t kpf = (size_t)(it + 2) * 32;
                __builtin_prefetch(gptr + kpf * MLP, 0, 3);
                __builtin_prefetch(uptr + kpf * MLP, 0, 3);
            }
            __syncthreads();                        // single barrier per K-step

            int fm = wm * 16 + l15, fn = wn * 16 + l15;
            union { uint4 q[2]; v16bf v; } fa, fg, fu;
            fa.q[0] = *(const uint4*)&pA[fm][h * 8];
            fa.q[1] = *(const uint4*)&pA[fm][h * 8 + 4];
            fg.q[0] = *(const uint4*)&pG[fn][h * 8];
            fg.q[1] = *(const uint4*)&pG[fn][h * 8 + 4];
            fu.q[0] = *(const uint4*)&pU[fn][h * 8];
            fu.q[1] = *(const uint4*)&pU[fn][h * 8 + 4];
            accG = __builtin_amdgcn_wmma_f32_16x16x32_bf16(false, fa.v, false, fg.v,
                                                           (short)0, accG, false, false);
            accU = __builtin_amdgcn_wmma_f32_16x16x32_bf16(false, fa.v, false, fu.v,
                                                           (short)0, accU, false, false);
        }
        int ncol = n0 + wn * 16 + l15;
        #pragma unroll
        for (int i = 0; i < 8; ++i) {
            int rsub = wm * 16 + i + 8 * h;
            int rg = mt * 32 + rsub;
            if (rg < count) {
                int r = base + rg;
                float wgt = rowW[r];
                float g = accG[i], u = accU[i];
                float sig = __builtin_amdgcn_rcpf(1.f + __expf(-g));
                act[(size_t)r * MLP + ncol] = f2bf(g * sig * u * wgt);
            }
        }
    }
}

// ---------------------------------------------------------------------------
// K5: grouped down-proj GEMM (bf16 WMMA). A-tile (already bf16) is staged with
//     CDNA5 async global->LDS copies (ASYNCcnt); B converted fp32->bf16 via
//     registers. Double-buffered, one barrier per K-step.
// ---------------------------------------------------------------------------
__global__ __launch_bounds__(256) void moe_down(
    const unsigned short* __restrict__ act, const float* __restrict__ downW,
    const int* __restrict__ offsets, const int* __restrict__ counts,
    const int* __restrict__ rowTok, float* __restrict__ out) {
    __shared__ __attribute__((aligned(16))) unsigned ldsA[2][32][20];
    __shared__ __attribute__((aligned(16))) unsigned ldsB[2][64][20];

    int e = blockIdx.y;
    int count = counts[e];
    if (count == 0) return;
    int base = offsets[e];
    int n0 = blockIdx.x * 64;

    int tid = threadIdx.x, lane = tid & 31, wid = tid >> 5;
    int wm = wid & 1, wn = wid >> 1;
    int h = lane >> 4, l15 = lane & 15;
    int arow = tid >> 3, ac = tid & 7;          // A: row, 8-byte chunk
    int kp = tid >> 4, n4 = (tid & 15) * 4;     // B: k-pair, n offset
    const int NT = MLP / 32;

    for (int mt = 0; mt * 32 < count; ++mt) {
        int aidx = mt * 32 + arow; if (aidx >= count) aidx = count - 1;
        const unsigned short* aptr = act + (size_t)(base + aidx) * MLP + ac * 4;
        const float* bptr = downW + ((size_t)e * MLP + 2 * kp) * HIDDEN + n0 + n4;

        unsigned lds0 = (unsigned)(uintptr_t)&ldsA[0][arow][ac * 2];
        unsigned lds1 = (unsigned)(uintptr_t)&ldsA[1][arow][ac * 2];

        // prologue: async A tile 0 -> LDS buf0, B tile 0 -> regs
        asm volatile("global_load_async_to_lds_b64 %0, %1, off"
                     :: "v"(lds0), "v"(aptr) : "memory");
        float4 d0 = *(const float4*)(bptr);
        float4 d1 = *(const float4*)(bptr + HIDDEN);

        v8f acc = {};
        for (int it = 0; it < NT; ++it) {
            unsigned (*pA)[20] = ldsA[it & 1];
            unsigned (*pB)[20] = ldsB[it & 1];
            pB[n4 + 0][kp] = pack2(d0.x, d1.x);
            pB[n4 + 1][kp] = pack2(d0.y, d1.y);
            pB[n4 + 2][kp] = pack2(d0.z, d1.z);
            pB[n4 + 3][kp] = pack2(d0.w, d1.w);
            if (it + 1 < NT) {
                size_t ko = (size_t)(it + 1) * 32;
                d0 = *(const float4*)(bptr + ko * HIDDEN);
                d1 = *(const float4*)(bptr + ko * HIDDEN + HIDDEN);
            }
            if (it + 2 < NT)
                __builtin_prefetch(bptr + (size_t)(it + 2) * 32 * HIDDEN, 0, 3);

            // my async A for buffer (it&1) must be done; barrier makes it
            // block-wide (every wave waited before arriving).
            asm volatile("s_wait_asynccnt 0x0" ::: "memory");
            __syncthreads();

            if (it + 1 < NT) {
                // safe only after the barrier: all reads of buf (it+1)&1
                // (compute at it-1) completed before every wave's barrier.
                unsigned ldsn = (it & 1) ? lds0 : lds1;
                asm volatile("global_load_async_to_lds_b64 %0, %1, off"
                             :: "v"(ldsn), "v"(aptr + (it + 1) * 32) : "memory");
            }

            int fm = wm * 16 + l15, fn = wn * 16 + l15;
            union { uint4 q[2]; v16bf v; } fa, fb;
            fa.q[0] = *(const uint4*)&pA[fm][h * 8];
            fa.q[1] = *(const uint4*)&pA[fm][h * 8 + 4];
            fb.q[0] = *(const uint4*)&pB[fn][h * 8];
            fb.q[1] = *(const uint4*)&pB[fn][h * 8 + 4];
            acc = __builtin_amdgcn_wmma_f32_16x16x32_bf16(false, fa.v, false, fb.v,
                                                          (short)0, acc, false, false);
        }
        int ncol = n0 + wn * 16 + l15;
        #pragma unroll
        for (int i = 0; i < 8; ++i) {
            int rsub = wm * 16 + i + 8 * h;
            int rg = mt * 32 + rsub;
            if (rg < count) {
                int rr = base + rg;
                atomicAdd(&out[(size_t)rowTok[rr] * HIDDEN + ncol], acc[i]);
            }
        }
    }
}

// ---------------------------------------------------------------------------
extern "C" void kernel_launch(void* const* d_in, const int* in_sizes, int n_in,
                              void* d_out, int out_size, void* d_ws, size_t ws_size,
                              hipStream_t stream) {
    const float* hs = (const float*)d_in[0];   // [1024,2048]
    const float* rw = (const float*)d_in[1];   // [2048,64]
    const float* gp = (const float*)d_in[2];   // [64,2048,768]
    const float* up = (const float*)d_in[3];   // [64,2048,768]
    const float* dp = (const float*)d_in[4];   // [64,768,2048]
    float* out = (float*)d_out;                // [1024,2048]

    char* ws = (char*)d_ws;
    int*   counts  = (int*)(ws + 0);
    int*   offsets = (int*)(ws + 256);
    int*   cursor  = (int*)(ws + 512);
    int*   selIdx  = (int*)(ws + 768);
    float* selW    = (float*)(ws + 768 + 32768);
    int*   rowTok  = (int*)(ws + 768 + 65536);
    float* rowW    = (float*)(ws + 768 + 98304);
    unsigned short* act = (unsigned short*)(ws + 131840);  // [8192,768] bf16

    hipMemsetAsync(counts, 0, 256, stream);
    hipMemsetAsync(out, 0, (size_t)TOKENS * HIDDEN * sizeof(float), stream);

    moe_router<<<TOKENS / 8, 256, 0, stream>>>(hs, rw, selIdx, selW, counts);
    moe_scan<<<1, 64, 0, stream>>>(counts, offsets, cursor);
    moe_scatter<<<TOKENS / 256, 256, 0, stream>>>(selIdx, selW, cursor, rowTok, rowW);
    moe_gateup<<<dim3(MLP / 64, NUM_EXPERTS), 256, 0, stream>>>(
        hs, gp, up, offsets, counts, rowTok, rowW, act);
    moe_down<<<dim3(HIDDEN / 64, NUM_EXPERTS), 256, 0, stream>>>(
        act, dp, offsets, counts, rowTok, out);
}
